// GroupMambaLayer_55078660603911
// MI455X (gfx1250) — matrix operand order repaired and checked by reading, not compiled, not run on hardware
//
#include <hip/hip_runtime.h>
#include <hip/hip_bf16.h>

// GroupMamba layer for MI455X (gfx1250, wave32).
// Pipeline: ln1 -> zmean -> gate -> in_proj(WMMA) -> dwconv3x3 -> x_dbl/delta
//        -> chunked assoc. scan (3 kernels) -> merge+LN128 -> out_proj(WMMA)
//        -> premix+LN512 -> proj(WMMA, transposed store).
// GEMMs: v_wmma_f32_16x16x32_f16, 128x32 block tile, fragment-major LDS.
// Tile staging is double-buffered with CDNA5 async global->LDS copies
// (GLOBAL_LOAD_ASYNC_TO_LDS_B128/B64 + s_wait_asynccnt) when the toolchain
// exposes the builtins; otherwise a register-prefetch software pipeline.

typedef __attribute__((ext_vector_type(16))) _Float16 v16h;
typedef __attribute__((ext_vector_type(8)))  _Float16 v8h;
typedef __attribute__((ext_vector_type(4)))  _Float16 v4h;
typedef __attribute__((ext_vector_type(8)))  float    v8f;

// async-copy builtin operand types (per hipcc diagnostic: vector-of-int
// pointers; b128 -> int4, b64 -> int2; global side is address_space(1))
typedef int i4v __attribute__((vector_size(16)));
typedef int i2v __attribute__((vector_size(8)));
#define ASG __attribute__((address_space(1)))
#define ASL __attribute__((address_space(3)))

#if __has_builtin(__builtin_amdgcn_global_load_async_to_lds_b128) && \
    __has_builtin(__builtin_amdgcn_global_load_async_to_lds_b64)  && \
    __has_builtin(__builtin_amdgcn_s_wait_asynccnt)
#define USE_ASYNC_LDS 1
#else
#define USE_ASYNC_LDS 0
#endif

#if USE_ASYNC_LDS
#define AS_SZ 8192   // double-buffered 128x32 f16 A tile
#define BS_SZ 2048   // double-buffered 32x32 f16 B tile
#else
#define AS_SZ 4096
#define BS_SZ 1024
#endif

#define GN  4
#define BB  8
#define CC  512
#define HH  64
#define WWD 64
#define LL  4096
#define BL  32768   // BB*LL
#define DD  128
#define RR  8
#define NCH 16      // scan chunks per sequence
#define CHL 256     // chunk length (NCH*CHL == LL)

__device__ __forceinline__ float siluf(float x) { return x / (1.f + __expf(-x)); }
__device__ __forceinline__ float sigmf(float x) { return 1.f / (1.f + __expf(-x)); }

// cross_scan index maps (g is 0-based here; reference g+1)
__device__ __forceinline__ int scan_to_spatial(int g, int s) {
  switch (g) {
    case 0: return s;
    case 1: { int h = s & (HH - 1); int w = s >> 6; return h * WWD + w; }
    case 2: return LL - 1 - s;
    default: { int j = LL - 1 - s; int h = j & (HH - 1); int w = j >> 6; return h * WWD + w; }
  }
}
__device__ __forceinline__ int spatial_to_scan(int g, int l) {
  switch (g) {
    case 0: return l;
    case 1: { int h = l >> 6; int w = l & (WWD - 1); return w * HH + h; }
    case 2: return LL - 1 - l;
    default: { int h = l >> 6; int w = l & (WWD - 1); return LL - 1 - (w * HH + h); }
  }
}

template <int NT>
__device__ __forceinline__ void block_reduce_2(float& a, float& b, float* red) {
  int t = threadIdx.x;
  red[t] = a; red[NT + t] = b;
  __syncthreads();
  #pragma unroll
  for (int off = NT / 2; off > 0; off >>= 1) {
    if (t < off) { red[t] += red[t + off]; red[NT + t] += red[NT + t + off]; }
    __syncthreads();
  }
  a = red[0]; b = red[NT];
}

// ---------------------------------------------------------------------------
// Block GEMM core: 128x32 output tile = A(M,K f16, stride lda) * W(N,K f16)^T.
// 256 threads = 8 waves; wave (wm 0..3, wn 0..1) computes M-tiles wm and wm+4
// of N-tile wn (2 WMMAs/wave/K-step, B fragment reused).
// LDS tiles are fragment-major:
//   A: frag element j of lane (hi,mr) holds A[mr][K], K = j + 8*[j>=8] + 8*hi
//   B: frag element j of lane (hi,mr) holds B[K][mr], K = j + 8*[j>=8] + 8*hi
// Each thread stages one 16-half A slice (two 16B groups -> slots hi=0/1) and
// one 4-half B slice; fragments are fetched as 2x ds_load_b128 per lane.
// ---------------------------------------------------------------------------
__device__ __forceinline__ void wmma_block_gemm2(
    const _Float16* __restrict__ A, int lda,
    const _Float16* __restrict__ W, int ldw,
    int K, int m0, int n0,
    _Float16* As, _Float16* Bs,
    v8f& acc0, v8f& acc1) {
  const int t = threadIdx.x;
  const int lane = t & 31, wave = t >> 5;
  const int wm = wave & 3, wn = wave >> 2;
  acc0 = (v8f){0.f, 0.f, 0.f, 0.f, 0.f, 0.f, 0.f, 0.f};
  acc1 = (v8f){0.f, 0.f, 0.f, 0.f, 0.f, 0.f, 0.f, 0.f};
  // staging decode (A: 128 rows x 32 K; thread owns 16 halves of one row)
  const int sm  = t >> 1;              // row 0..127
  const int skk = (t & 1) << 4;        // K base 0 or 16
  const int smt = sm >> 4, smr = sm & 15;
  // staging decode (B: 32 rows x 32 K; thread owns 4 halves)
  const int bn  = t >> 3;              // n 0..31
  const int bkk = (t & 7) << 2;        // K base 0..28
  const int bhi = (bkk >> 3) & 1;
  const int bjb = (bkk & 7) | ((bkk >> 4) << 3);
  const int bnt = bn >> 4, bnr = bn & 15;
  // LDS fragment slots (halves): kkk=skk -> hi 0, kkk=skk+8 -> hi 1, jb=skk/2
  const int jbA    = (skk >> 4) << 3;
  const int slotA0 = ((smt * 32 + smr) << 4) + jbA;
  const int slotA1 = ((smt * 32 + 16 + smr) << 4) + jbA;
  const int slotB  = ((bnt * 32 + bhi * 16 + bnr) << 4) + bjb;
  const _Float16* sa = A + (size_t)(m0 + sm) * lda + skk;
  const _Float16* sb = W + (size_t)(n0 + bn) * ldw + bkk;

#if USE_ASYNC_LDS
  // Double-buffered async global->LDS scatter copies (ASYNCcnt-tracked).
  int p = 0;
  {
    __builtin_amdgcn_global_load_async_to_lds_b128(
        (ASG i4v*)sa, (ASL i4v*)&As[slotA0], 0, 0);
    __builtin_amdgcn_global_load_async_to_lds_b128(
        (ASG i4v*)(sa + 8), (ASL i4v*)&As[slotA1], 0, 0);
    __builtin_amdgcn_global_load_async_to_lds_b64(
        (ASG i2v*)sb, (ASL i2v*)&Bs[slotB], 0, 0);
  }
  for (int k0 = 0; k0 < K; k0 += 32) {
    const bool more = (k0 + 32) < K;
    if (more) {  // issue next tile into the other buffer; overlaps with WMMA
      int q = p ^ 1;
      __builtin_amdgcn_global_load_async_to_lds_b128(
          (ASG i4v*)(sa + k0 + 32), (ASL i4v*)&As[q * 4096 + slotA0], 0, 0);
      __builtin_amdgcn_global_load_async_to_lds_b128(
          (ASG i4v*)(sa + k0 + 40), (ASL i4v*)&As[q * 4096 + slotA1], 0, 0);
      __builtin_amdgcn_global_load_async_to_lds_b64(
          (ASG i2v*)(sb + k0 + 32), (ASL i2v*)&Bs[q * 1024 + slotB], 0, 0);
      __builtin_amdgcn_s_wait_asynccnt(3);  // oldest 3 (this tile) retired
    } else {
      __builtin_amdgcn_s_wait_asynccnt(0);
    }
    __syncthreads();
    const _Float16* Ap = As + p * 4096;
    const _Float16* Bp = Bs + p * 1024;
    v16h bf = *(const v16h*)&Bp[(wn * 32 + lane) << 4];
    v16h a0 = *(const v16h*)&Ap[(wm * 32 + lane) << 4];
    v16h a1 = *(const v16h*)&Ap[((wm + 4) * 32 + lane) << 4];
    acc0 = __builtin_amdgcn_wmma_f32_16x16x32_f16(false, a0, false, bf, (short)0,
                                                  acc0, false, false);
    acc1 = __builtin_amdgcn_wmma_f32_16x16x32_f16(false, a1, false, bf, (short)0,
                                                  acc1, false, false);
    __syncthreads();
    p ^= 1;
  }
#else
  // Register-prefetch software pipeline: next tile's global loads issue while
  // this tile's fragments/WMMAs execute (waitcnt lands at next ds_store).
  v8h ra0 = *(const v8h*)sa;
  v8h ra1 = *(const v8h*)(sa + 8);
  v4h rb  = *(const v4h*)sb;
  for (int k0 = 0; k0 < K; k0 += 32) {
    __syncthreads();
    *(v8h*)&As[slotA0] = ra0;
    *(v8h*)&As[slotA1] = ra1;
    *(v4h*)&Bs[slotB]  = rb;
    if (k0 + 32 < K) {
      ra0 = *(const v8h*)(sa + k0 + 32);
      ra1 = *(const v8h*)(sa + k0 + 40);
      rb  = *(const v4h*)(sb + k0 + 32);
    }
    __syncthreads();
    v16h bf = *(const v16h*)&Bs[(wn * 32 + lane) << 4];
    v16h a0 = *(const v16h*)&As[(wm * 32 + lane) << 4];
    v16h a1 = *(const v16h*)&As[((wm + 4) * 32 + lane) << 4];
    acc0 = __builtin_amdgcn_wmma_f32_16x16x32_f16(false, a0, false, bf, (short)0,
                                                  acc0, false, false);
    acc1 = __builtin_amdgcn_wmma_f32_16x16x32_f16(false, a1, false, bf, (short)0,
                                                  acc1, false, false);
  }
#endif
}

// ---------------------------------------------------------------------------
__global__ void k_cvt(const float* __restrict__ s, _Float16* __restrict__ d, int n) {
  int i = blockIdx.x * 256 + threadIdx.x;
  if (i < n) d[i] = (_Float16)s[i];
}

// LayerNorm over C: x (B,C,L) -> xn (B,L,C) f32 + f16
__global__ __launch_bounds__(256) void k_ln1(
    const float* __restrict__ x, const float* __restrict__ lng,
    const float* __restrict__ lnb, float* __restrict__ xn,
    _Float16* __restrict__ xnh) {
  __shared__ float red[512];
  int m = blockIdx.x;
  int b = m >> 12, l = m & (LL - 1);
  int t = threadIdx.x;
  const float* xp = x + (size_t)b * CC * LL + l;
  float v0 = xp[(size_t)t * LL];
  float v1 = xp[(size_t)(t + 256) * LL];
  float s = v0 + v1, ss = v0 * v0 + v1 * v1;
  block_reduce_2<256>(s, ss, red);
  float mu = s * (1.f / CC);
  float var = ss * (1.f / CC) - mu * mu;
  float rstd = rsqrtf(var + 1e-5f);
  size_t o = (size_t)m * CC;
  float y0 = (v0 - mu) * rstd * lng[t] + lnb[t];
  float y1 = (v1 - mu) * rstd * lng[t + 256] + lnb[t + 256];
  xn[o + t] = y0;            xn[o + t + 256] = y1;
  xnh[o + t] = (_Float16)y0; xnh[o + t + 256] = (_Float16)y1;
}

// z[b,c] = mean_l xn[b,l,c]
__global__ __launch_bounds__(256) void k_zmean(const float* __restrict__ xn,
                                               float* __restrict__ z) {
  __shared__ float red[256];
  int idx = blockIdx.x;                 // b*CC + c
  int b = idx >> 9, c = idx & (CC - 1);
  int t = threadIdx.x;
  const float* p = xn + (size_t)b * LL * CC + c;
  float s = 0.f;
  for (int l = t; l < LL; l += 256) s += p[(size_t)l * CC];
  red[t] = s; __syncthreads();
  for (int off = 128; off > 0; off >>= 1) {
    if (t < off) red[t] += red[t + off];
    __syncthreads();
  }
  if (t == 0) z[idx] = red[0] * (1.f / LL);
}

// gate = sigmoid(relu(z @ fc1^T + b1) @ fc2^T + b2); one block per batch
__global__ __launch_bounds__(256) void k_gate(
    const float* __restrict__ z, const float* __restrict__ w1,
    const float* __restrict__ b1, const float* __restrict__ w2,
    const float* __restrict__ b2, float* __restrict__ gate) {
  __shared__ float hrow[32];
  int b = blockIdx.x, t = threadIdx.x;
  if (t < 32) {
    float a = b1[t];
    const float* wp = w1 + (size_t)t * CC;
    const float* zp = z + (size_t)b * CC;
    for (int c = 0; c < CC; ++c) a += wp[c] * zp[c];
    hrow[t] = fmaxf(a, 0.f);
  }
  __syncthreads();
  for (int c = t; c < CC; c += 256) {
    float a = b2[c];
    const float* wp = w2 + (size_t)c * 32;
    #pragma unroll
    for (int r = 0; r < 32; ++r) a += hrow[r] * wp[r];
    gate[b * CC + c] = sigmf(a);
  }
}

// in_proj: xz = xn_chunk(g) @ W^T ; first half -> xd, second half -> silu -> zbuf
__global__ __launch_bounds__(256) void k_inproj(
    const _Float16* __restrict__ xnh, const _Float16* __restrict__ wh,
    float* __restrict__ xd, float* __restrict__ zbuf) {
  __shared__ __align__(32) _Float16 As[AS_SZ];
  __shared__ __align__(32) _Float16 Bs[BS_SZ];
  int g = blockIdx.z;
  int m0 = blockIdx.x * 128, n0 = blockIdx.y * 32;
  v8f acc0, acc1;
  wmma_block_gemm2(xnh + g * DD, CC, wh + (size_t)g * 256 * DD, DD, DD,
                   m0, n0, As, Bs, acc0, acc1);
  int lane = threadIdx.x & 31, wave = threadIdx.x >> 5;
  int wm = wave & 3, wn = wave >> 2;
  bool hi = lane >= 16;
  int mr = lane & 15;
  int col = n0 + wn * 16 + mr;
  size_t gb = (size_t)g * BL;
  #pragma unroll
  for (int v = 0; v < 8; ++v) {
    int row0 = m0 + wm * 16 + v + (hi ? 8 : 0);
    int row1 = row0 + 64;
    float x0 = acc0[v], x1 = acc1[v];
    if (col < DD) {
      xd[(gb + row0) * DD + col] = x0;
      xd[(gb + row1) * DD + col] = x1;
    } else {
      zbuf[(gb + row0) * DD + (col - DD)] = siluf(x0);
      zbuf[(gb + row1) * DD + (col - DD)] = siluf(x1);
    }
  }
}

// depthwise 3x3 SAME (correlation) + bias + silu, layout (g,b,l,d)
__global__ __launch_bounds__(256) void k_conv(
    const float* __restrict__ xd, const float* __restrict__ cw,
    const float* __restrict__ cb, float* __restrict__ xc) {
  size_t i = (size_t)blockIdx.x * 256 + threadIdx.x;   // over GN*BL*DD
  int d = (int)(i & (DD - 1));
  size_t p = i >> 7;                                   // g*BL + b*LL + l
  int l = (int)(p & (LL - 1));
  int gb = (int)(p >> 12);
  int g = gb >> 3;
  int h = l >> 6, w = l & (WWD - 1);
  const float* base = xd + (p - l) * DD + d;
  const float* wgt  = cw + (size_t)(g * DD + d) * 9;
  float acc = cb[g * DD + d];
  #pragma unroll
  for (int ky = 0; ky < 3; ++ky) {
    int h2 = h + ky - 1;
    if (h2 < 0 || h2 >= HH) continue;
    #pragma unroll
    for (int kx = 0; kx < 3; ++kx) {
      int w2 = w + kx - 1;
      if (w2 < 0 || w2 >= WWD) continue;
      acc += base[(size_t)(h2 * WWD + w2) * DD] * wgt[ky * 3 + kx];
    }
  }
  xc[i] = siluf(acc);
}

// per scan position: x_dbl -> delta -> dA, dBu, C  (block = 128 threads over d)
__global__ __launch_bounds__(128) void k_xdbl(
    const float* __restrict__ xc, const float* __restrict__ xpw,
    const float* __restrict__ dtpw, const float* __restrict__ dtpb,
    const float* __restrict__ alog, float* __restrict__ dA,
    float* __restrict__ dBu, float* __restrict__ Cms) {
  __shared__ float su[DD];
  __shared__ float sd[10];
  int p = blockIdx.x;                 // gb*LL + s
  int s = p & (LL - 1);
  int gb = p >> 12;
  int g = gb >> 3;
  int d = threadIdx.x;
  int l = scan_to_spatial(g, s);
  size_t pb = (size_t)gb * LL;
  float u = xc[(pb + l) * DD + d];
  su[d] = u;
  __syncthreads();
  if (d < 10) {
    const float* wp = xpw + ((size_t)g * 10 + d) * DD;
    float a = 0.f;
    for (int j = 0; j < DD; ++j) a += wp[j] * su[j];
    sd[d] = a;
  }
  __syncthreads();
  float dt = dtpb[g * DD + d];
  const float* dw = dtpw + (size_t)(g * DD + d) * RR;
  #pragma unroll
  for (int r = 0; r < RR; ++r) dt += sd[r] * dw[r];
  float delta = (dt > 20.f) ? dt : log1pf(__expf(dt));
  float A = -__expf(alog[g * DD + d]);       // N == 1
  size_t o = (pb + s) * DD + d;
  dA[o]  = __expf(delta * A);
  dBu[o] = delta * u * sd[8];
  if (d == 0) Cms[pb + s] = sd[9];
}

// chunked associative scan of h' = a*h + b
__global__ __launch_bounds__(256) void k_scanA(
    const float* __restrict__ dA, const float* __restrict__ dBu,
    float* __restrict__ aprod, float* __restrict__ bsum) {
  int i = blockIdx.x * 256 + threadIdx.x;   // over GN*BB*NCH*DD
  int d = i & (DD - 1);
  int r = i >> 7;
  int ch = r & (NCH - 1);
  int gb = r >> 4;
  size_t pb = (size_t)gb * LL + (size_t)ch * CHL;
  float ap = 1.f, h = 0.f;
  for (int s = 0; s < CHL; ++s) {
    size_t o = (pb + s) * DD + d;
    if ((s & 31) == 0 && s + 32 < CHL) {   // stream ahead (global_prefetch_b8)
      __builtin_prefetch(dA + o + (size_t)32 * DD, 0, 1);
      __builtin_prefetch(dBu + o + (size_t)32 * DD, 0, 1);
    }
    float a = dA[o];
    ap *= a;
    h = a * h + dBu[o];
  }
  size_t oo = ((size_t)gb * NCH + ch) * DD + d;
  aprod[oo] = ap; bsum[oo] = h;
}

__global__ __launch_bounds__(256) void k_scanB(
    const float* __restrict__ aprod, const float* __restrict__ bsum,
    float* __restrict__ h0) {
  int i = blockIdx.x * 256 + threadIdx.x;   // over GN*BB*DD
  int d = i & (DD - 1);
  int gb = i >> 7;
  float h = 0.f;
  for (int ch = 0; ch < NCH; ++ch) {
    size_t oo = ((size_t)gb * NCH + ch) * DD + d;
    h0[oo] = h;
    h = aprod[oo] * h + bsum[oo];
  }
}

__global__ __launch_bounds__(256) void k_scanC(
    const float* __restrict__ dA, const float* __restrict__ dBu,
    const float* __restrict__ Cms, const float* __restrict__ h0,
    float* __restrict__ ys) {
  int i = blockIdx.x * 256 + threadIdx.x;
  int d = i & (DD - 1);
  int r = i >> 7;
  int ch = r & (NCH - 1);
  int gb = r >> 4;
  size_t pb = (size_t)gb * LL + (size_t)ch * CHL;
  float h = h0[((size_t)gb * NCH + ch) * DD + d];
  for (int s = 0; s < CHL; ++s) {
    size_t o = (pb + s) * DD + d;
    if ((s & 31) == 0 && s + 32 < CHL) {
      __builtin_prefetch(dA + o + (size_t)32 * DD, 0, 1);
      __builtin_prefetch(dBu + o + (size_t)32 * DD, 0, 1);
    }
    h = dA[o] * h + dBu[o];
    ys[o] = h * Cms[pb + s];
  }
}

// cross_merge + Ds*u + LayerNorm(128) + * silu(z)  -> f16
__global__ __launch_bounds__(128) void k_merge(
    const float* __restrict__ ys, const float* __restrict__ xc,
    const float* __restrict__ Ds, const float* __restrict__ og,
    const float* __restrict__ ob, const float* __restrict__ zbuf,
    _Float16* __restrict__ vh) {
  __shared__ float red[2 * DD];
  int p = blockIdx.x;                 // gb*LL + l (spatial)
  int l = p & (LL - 1);
  int gb = p >> 12;
  int g = gb >> 3;
  int d = threadIdx.x;
  int s = spatial_to_scan(g, l);
  size_t pb = (size_t)gb * LL;
  float y = ys[(pb + s) * DD + d] + Ds[g * DD + d] * xc[(pb + l) * DD + d];
  float a = y, b = y * y;
  block_reduce_2<DD>(a, b, red);
  float mu = a * (1.f / DD);
  float var = b * (1.f / DD) - mu * mu;
  float rstd = rsqrtf(var + 1e-5f);
  float yn = (y - mu) * rstd * og[g * DD + d] + ob[g * DD + d];
  vh[(pb + l) * DD + d] = (_Float16)(yn * zbuf[(pb + l) * DD + d]);
}

// out_proj per group: ybuf[b,l, g*128+n] = v @ W^T
__global__ __launch_bounds__(256) void k_outproj(
    const _Float16* __restrict__ vh, const _Float16* __restrict__ wh,
    float* __restrict__ ybuf) {
  __shared__ __align__(32) _Float16 As[AS_SZ];
  __shared__ __align__(32) _Float16 Bs[BS_SZ];
  int g = blockIdx.z;
  int m0 = blockIdx.x * 128, n0 = blockIdx.y * 32;
  v8f acc0, acc1;
  wmma_block_gemm2(vh + (size_t)g * BL * DD, DD, wh + (size_t)g * DD * DD, DD, DD,
                   m0, n0, As, Bs, acc0, acc1);
  int lane = threadIdx.x & 31, wave = threadIdx.x >> 5;
  int wm = wave & 3, wn = wave >> 2;
  bool hi = lane >= 16;
  int mr = lane & 15;
  int col = n0 + wn * 16 + mr;
  #pragma unroll
  for (int v = 0; v < 8; ++v) {
    int row0 = m0 + wm * 16 + v + (hi ? 8 : 0);
    int row1 = row0 + 64;
    ybuf[(size_t)row0 * CC + g * DD + col] = acc0[v];
    ybuf[(size_t)row1 * CC + g * DD + col] = acc1[v];
  }
}

// xm = ybuf * skip * xn * gate ; LayerNorm(512) -> f16
__global__ __launch_bounds__(256) void k_premix(
    const float* __restrict__ ybuf, const float* __restrict__ xn,
    const float* __restrict__ gate, const float* __restrict__ skip,
    const float* __restrict__ lng, const float* __restrict__ lnb,
    _Float16* __restrict__ xmh) {
  __shared__ float red[512];
  int m = blockIdx.x;
  int b = m >> 12;
  int t = threadIdx.x;
  float sk = skip[0];
  size_t o = (size_t)m * CC;
  float v0 = ybuf[o + t]       * sk * xn[o + t]       * gate[b * CC + t];
  float v1 = ybuf[o + t + 256] * sk * xn[o + t + 256] * gate[b * CC + t + 256];
  float s = v0 + v1, ss = v0 * v0 + v1 * v1;
  block_reduce_2<256>(s, ss, red);
  float mu = s * (1.f / CC);
  float var = ss * (1.f / CC) - mu * mu;
  float rstd = rsqrtf(var + 1e-5f);
  xmh[o + t]       = (_Float16)((v0 - mu) * rstd * lng[t]       + lnb[t]);
  xmh[o + t + 256] = (_Float16)((v1 - mu) * rstd * lng[t + 256] + lnb[t + 256]);
}

// final proj, store transposed to (B,C,L)
__global__ __launch_bounds__(256) void k_proj(
    const _Float16* __restrict__ xmh, const _Float16* __restrict__ wh,
    const float* __restrict__ bias, float* __restrict__ out) {
  __shared__ __align__(32) _Float16 As[AS_SZ];
  __shared__ __align__(32) _Float16 Bs[BS_SZ];
  int m0 = blockIdx.x * 128, n0 = blockIdx.y * 32;
  v8f acc0, acc1;
  wmma_block_gemm2(xmh, CC, wh, CC, CC, m0, n0, As, Bs, acc0, acc1);
  int lane = threadIdx.x & 31, wave = threadIdx.x >> 5;
  int wm = wave & 3, wn = wave >> 2;
  bool hi = lane >= 16;
  int mr = lane & 15;
  int col = n0 + wn * 16 + mr;
  float bv = bias[col];
  #pragma unroll
  for (int v = 0; v < 8; ++v) {
    int row0 = m0 + wm * 16 + v + (hi ? 8 : 0);
    int row1 = row0 + 64;
    int b0 = row0 >> 12, l0 = row0 & (LL - 1);
    int b1 = row1 >> 12, l1 = row1 & (LL - 1);
    out[((size_t)b0 * CC + col) * LL + l0] = acc0[v] + bv;
    out[((size_t)b1 * CC + col) * LL + l1] = acc1[v] + bv;
  }
}

// ---------------------------------------------------------------------------
extern "C" void kernel_launch(void* const* d_in, const int* in_sizes, int n_in,
                              void* d_out, int out_size, void* d_ws, size_t ws_size,
                              hipStream_t stream) {
  (void)in_sizes; (void)n_in; (void)out_size; (void)ws_size;
  const float* x     = (const float*)d_in[0];
  const float* ln_g  = (const float*)d_in[1];
  const float* ln_b  = (const float*)d_in[2];
  const float* fc1_w = (const float*)d_in[3];
  const float* fc1_b = (const float*)d_in[4];
  const float* fc2_w = (const float*)d_in[5];
  const float* fc2_b = (const float*)d_in[6];
  const float* inpw  = (const float*)d_in[7];
  const float* convw = (const float*)d_in[8];
  const float* convb = (const float*)d_in[9];
  const float* xpw   = (const float*)d_in[10];
  const float* dtpw  = (const float*)d_in[11];
  const float* dtpb  = (const float*)d_in[12];
  const float* alog  = (const float*)d_in[13];
  const float* Dsp   = (const float*)d_in[14];
  const float* og    = (const float*)d_in[15];
  const float* ob    = (const float*)d_in[16];
  const float* outw  = (const float*)d_in[17];
  const float* projw = (const float*)d_in[18];
  const float* projb = (const float*)d_in[19];
  const float* skip  = (const float*)d_in[20];
  float* out = (float*)d_out;

  char* ws = (char*)d_ws;
  size_t off = 0;
  auto take = [&](size_t bytes) -> char* {
    char* p = ws + off;
    off += (bytes + 255) & ~(size_t)255;
    return p;
  };
  // ~418 MiB total; buffers aliased across dead live-ranges
  float*    xn32 = (float*)take((size_t)BL * CC * 4);
  float*    zbuf = (float*)take((size_t)GN * BL * DD * 4);
  float*    xc   = (float*)take((size_t)GN * BL * DD * 4);
  float*    dBu  = (float*)take((size_t)GN * BL * DD * 4);  float* ybuf = dBu;
  float*    dAa  = (float*)take((size_t)GN * BL * DD * 4);  _Float16* xmh = (_Float16*)dAa;
  float*    xd   = (float*)take((size_t)GN * BL * DD * 4);  float* ys = xd;
  _Float16* xnh  = (_Float16*)take((size_t)BL * CC * 2);    _Float16* vh = xnh;
  float*    Cms  = (float*)take((size_t)GN * BL * 4);
  float*    zm   = (float*)take((size_t)BB * CC * 4);
  float*    gate = (float*)take((size_t)BB * CC * 4);
  _Float16* whIn = (_Float16*)take((size_t)GN * 256 * DD * 2);
  _Float16* whOu = (_Float16*)take((size_t)GN * DD * DD * 2);
  _Float16* whPj = (_Float16*)take((size_t)CC * CC * 2);
  float*    aprd = (float*)take((size_t)GN * BB * NCH * DD * 4);
  float*    bsum = (float*)take((size_t)GN * BB * NCH * DD * 4);
  float*    h0   = (float*)take((size_t)GN * BB * NCH * DD * 4);

  k_cvt<<<(GN * 256 * DD + 255) / 256, 256, 0, stream>>>(inpw, whIn, GN * 256 * DD);
  k_cvt<<<(GN * DD * DD + 255) / 256, 256, 0, stream>>>(outw, whOu, GN * DD * DD);
  k_cvt<<<(CC * CC + 255) / 256, 256, 0, stream>>>(projw, whPj, CC * CC);

  k_ln1<<<BL, 256, 0, stream>>>(x, ln_g, ln_b, xn32, xnh);
  k_zmean<<<BB * CC, 256, 0, stream>>>(xn32, zm);
  k_gate<<<BB, 256, 0, stream>>>(zm, fc1_w, fc1_b, fc2_w, fc2_b, gate);

  k_inproj<<<dim3(BL / 128, 256 / 32, GN), 256, 0, stream>>>(xnh, whIn, xd, zbuf);
  k_conv<<<(GN * BL * DD) / 256, 256, 0, stream>>>(xd, convw, convb, xc);
  k_xdbl<<<GN * BL, DD, 0, stream>>>(xc, xpw, dtpw, dtpb, alog, dAa, dBu, Cms);

  k_scanA<<<(GN * BB * NCH * DD) / 256, 256, 0, stream>>>(dAa, dBu, aprd, bsum);
  k_scanB<<<(GN * BB * DD) / 256, 256, 0, stream>>>(aprd, bsum, h0);
  k_scanC<<<(GN * BB * NCH * DD) / 256, 256, 0, stream>>>(dAa, dBu, Cms, h0, ys);

  k_merge<<<GN * BL, DD, 0, stream>>>(ys, xc, Dsp, og, ob, zbuf, vh);
  k_outproj<<<dim3(BL / 128, 128 / 32, GN), 256, 0, stream>>>(vh, whOu, ybuf);
  k_premix<<<BL, 256, 0, stream>>>(ybuf, xn32, gate, skip, ln_g, ln_b, xmh);
  k_proj<<<dim3(BL / 128, CC / 32), 256, 0, stream>>>(xmh, whPj, projb, out);
}
